// AnomalyAttention_51840255263060
// MI455X (gfx1250) — matrix operand order repaired and checked by reading, not compiled, not run on hardware
//
#include <hip/hip_runtime.h>
#include <math.h>

// AnomalyAttention for B=16, L=512, H=8, E=D=64 on gfx1250 (wave32, WMMA).
// Memory-bound (~0.54 GB @ 23.3 TB/s); matmuls on v_wmma_f32_16x16x32_f16,
// fp32 accumulation, all normalizations folded into per-row scalars.

typedef __attribute__((ext_vector_type(16))) _Float16 v16h;
typedef __attribute__((ext_vector_type(8)))  float    v8f;

#define Lseq 512
#define Hh   8
#define Ee   64
#define LDSTR 72   // padded LDS row stride (f16 elements) to dodge bank conflicts

// A fragment (16x32 f16): lane M = lane&15; K contiguous:
//  lanes<16 hold K=0..15, lanes>=16 hold K=16..31
__device__ __forceinline__ v16h ld_a(const _Float16* p, int row, int k0,
                                     int lane15, int laneHalf) {
  v16h a;
#pragma unroll
  for (int e = 0; e < 16; ++e) {
    int k = (laneHalf << 4) + e;
    a[e] = p[row * LDSTR + k0 + k];
  }
  return a;
}

// B fragment (32x16 f16) where B[k][n] = K[sBase+n][k0+k] (K^T for scores):
//  lane N = lane&15; lanes<16 hold K=0..15, lanes>=16 hold K=16..31
__device__ __forceinline__ v16h ld_bT(const _Float16* p, int sBase, int k0,
                                      int lane15, int laneHalf) {
  v16h bf;
#pragma unroll
  for (int e = 0; e < 16; ++e)
    bf[e] = p[(sBase + lane15) * LDSTR + k0 + (laneHalf << 4) + e];
  return bf;
}

// B fragment where B[k][n] = V[k0+k][dBase+n]  (fused @ V)
__device__ __forceinline__ v16h ld_bV(const _Float16* p, int dBase, int k0,
                                      int lane15, int laneHalf) {
  v16h bf;
#pragma unroll
  for (int e = 0; e < 16; ++e)
    bf[e] = p[(k0 + (laneHalf << 4) + e) * LDSTR + dBase + lane15];
  return bf;
}

__global__ __launch_bounds__(256)
void anomaly_attn_kernel(const float* __restrict__ Qg,
                         const float* __restrict__ Kg,
                         const float* __restrict__ Vg,
                         const float* __restrict__ Sg,
                         const float* __restrict__ Gg,
                         float* __restrict__ Og) {
  __shared__ _Float16 Qls[64 * LDSTR];
  __shared__ _Float16 Kls[64 * LDSTR];
  __shared__ _Float16 Vls[64 * LDSTR];
  __shared__ _Float16 Fls[64 * LDSTR];
  __shared__ float rowExp[64], rowPsum[64], negA[64], coefA[64], aEr[64], aPr[64];

  const int tid      = threadIdx.x;
  const int lane     = tid & 31;
  const int lane15   = lane & 15;
  const int laneHalf = lane >> 4;
  const int w        = tid >> 5;   // wave 0..7
  const int r        = w >> 1;     // 16-row sub-tile 0..3
  const int q        = w & 1;      // column-half selector

  const int row0 = blockIdx.x * 64;
  const int h    = blockIdx.y;
  const int b    = blockIdx.z;

  // ---- stage Q tile (64x64) fp32 -> f16 ----
  {
    int rr = tid >> 2;
    int c0 = (tid & 3) << 4;
    const float4* src =
        (const float4*)(Qg + (((size_t)b * Lseq + row0 + rr) * Hh + h) * Ee + c0);
#pragma unroll
    for (int j = 0; j < 4; ++j) {
      float4 v = src[j];
      Qls[rr * LDSTR + c0 + 4 * j + 0] = (_Float16)v.x;
      Qls[rr * LDSTR + c0 + 4 * j + 1] = (_Float16)v.y;
      Qls[rr * LDSTR + c0 + 4 * j + 2] = (_Float16)v.z;
      Qls[rr * LDSTR + c0 + 4 * j + 3] = (_Float16)v.w;
    }
  }

  // ---- per-row sigma-derived constants ----
  if (tid < 64) {
    rowExp[tid]  = 0.f;
    rowPsum[tid] = 0.f;
    float sraw = Sg[((size_t)b * Lseq + row0 + tid) * Hh + h];
    float sg   = 1.f / (1.f + __expf(-5.f * sraw)) + 1e-5f;
    float sig  = __expf(sg * 1.0986122886681098f) - 1.f;  // 3^sg - 1
    negA[tid]  = -1.f / (2.f * sig * sig);
    coefA[tid] = 0.3989422804014327f / sig;               // 1/(sqrt(2pi)*sig)
  }
  __syncthreads();

  // ---- prior row sums (analytic, full row, unmasked) ----
  {
    int rr = tid & 63;
    int part = tid >> 6;
    int l = row0 + rr;
    float ng = negA[rr];
    float acc = 0.f;
    for (int s = part * 128; s < part * 128 + 128; ++s) {
      float d = (float)(l - s);
      acc += __expf(ng * d * d);
    }
    atomicAdd(&rowPsum[rr], coefA[rr] * acc);
  }

  // ---- pass 1: softmax denominators (causal chunks only) ----
  float expAcc[8];
#pragma unroll
  for (int i = 0; i < 8; ++i) expAcc[i] = 0.f;

  for (int c = 0; c <= (int)blockIdx.x; ++c) {
    __syncthreads();
    {
      int rr = tid >> 2;
      int c0 = (tid & 3) << 4;
      const float4* src =
          (const float4*)(Kg + (((size_t)b * Lseq + c * 64 + rr) * Hh + h) * Ee + c0);
#pragma unroll
      for (int j = 0; j < 4; ++j) {
        float4 v = src[j];
        Kls[rr * LDSTR + c0 + 4 * j + 0] = (_Float16)v.x;
        Kls[rr * LDSTR + c0 + 4 * j + 1] = (_Float16)v.y;
        Kls[rr * LDSTR + c0 + 4 * j + 2] = (_Float16)v.z;
        Kls[rr * LDSTR + c0 + 4 * j + 3] = (_Float16)v.w;
      }
    }
    __syncthreads();
#pragma unroll
    for (int tt = 0; tt < 2; ++tt) {
      int tileCol = (2 * q + tt) * 16;
      v8f acc = {0, 0, 0, 0, 0, 0, 0, 0};
#pragma unroll
      for (int e0 = 0; e0 < 64; e0 += 32) {
        v16h a  = ld_a(Qls, 16 * r + lane15, e0, lane15, laneHalf);
        v16h bb = ld_bT(Kls, tileCol, e0, lane15, laneHalf);
        acc = __builtin_amdgcn_wmma_f32_16x16x32_f16(false, a, false, bb,
                                                     (short)0, acc, false, false);
      }
#pragma unroll
      for (int i = 0; i < 8; ++i) {
        int l = row0 + 16 * r + i + 8 * laneHalf;
        int s = c * 64 + tileCol + lane15;
        expAcc[i] += (s <= l) ? __expf(0.125f * acc[i]) : 0.f;
      }
    }
  }
#pragma unroll
  for (int i = 0; i < 8; ++i) {
    float v = expAcc[i];
#pragma unroll
    for (int off = 1; off < 16; off <<= 1) v += __shfl_xor(v, off, 32);
    if (lane15 == 0) atomicAdd(&rowExp[16 * r + i + 8 * laneHalf], v);
  }
  __syncthreads();

  // ---- fold all per-row normalizations into two scalars ----
  if (tid < 64) {
    float g   = 1.f / (1.f + __expf(-Gg[h]));
    float ps  = rowPsum[tid];
    float fs  = g + (1.f - g) * ps / (ps + 1e-8f);   // fused row sum
    float scl = 1.f / (fs + 1e-8f);
    aEr[tid] = scl * g / rowExp[tid];
    aPr[tid] = scl * (1.f - g) * coefA[tid] / (ps + 1e-8f);
  }
  __syncthreads();

  // ---- pass 2: fused = aE*exp(score/8) + aP*exp(-d^2/(2s^2)); out = fused @ V ----
  v8f outAcc[2];
#pragma unroll
  for (int tt = 0; tt < 2; ++tt) outAcc[tt] = (v8f){0, 0, 0, 0, 0, 0, 0, 0};

  for (int c = 0; c < 8; ++c) {
    __syncthreads();
    {
      int rr = tid >> 2;
      int c0 = (tid & 3) << 4;
      const float4* srcK =
          (const float4*)(Kg + (((size_t)b * Lseq + c * 64 + rr) * Hh + h) * Ee + c0);
      const float4* srcV =
          (const float4*)(Vg + (((size_t)b * Lseq + c * 64 + rr) * Hh + h) * Ee + c0);
#pragma unroll
      for (int j = 0; j < 4; ++j) {
        float4 vk = srcK[j];
        float4 vv = srcV[j];
        Kls[rr * LDSTR + c0 + 4 * j + 0] = (_Float16)vk.x;
        Kls[rr * LDSTR + c0 + 4 * j + 1] = (_Float16)vk.y;
        Kls[rr * LDSTR + c0 + 4 * j + 2] = (_Float16)vk.z;
        Kls[rr * LDSTR + c0 + 4 * j + 3] = (_Float16)vk.w;
        Vls[rr * LDSTR + c0 + 4 * j + 0] = (_Float16)vv.x;
        Vls[rr * LDSTR + c0 + 4 * j + 1] = (_Float16)vv.y;
        Vls[rr * LDSTR + c0 + 4 * j + 2] = (_Float16)vv.z;
        Vls[rr * LDSTR + c0 + 4 * j + 3] = (_Float16)vv.w;
      }
    }
    __syncthreads();

    const bool liveScores = (c <= (int)blockIdx.x);
#pragma unroll
    for (int tt = 0; tt < 2; ++tt) {
      int tileCol = (2 * q + tt) * 16;
      v8f acc = {0, 0, 0, 0, 0, 0, 0, 0};
      if (liveScores) {
#pragma unroll
        for (int e0 = 0; e0 < 64; e0 += 32) {
          v16h a  = ld_a(Qls, 16 * r + lane15, e0, lane15, laneHalf);
          v16h bb = ld_bT(Kls, tileCol, e0, lane15, laneHalf);
          acc = __builtin_amdgcn_wmma_f32_16x16x32_f16(false, a, false, bb,
                                                       (short)0, acc, false, false);
        }
      }
#pragma unroll
      for (int i = 0; i < 8; ++i) {
        int rr64 = 16 * r + i + 8 * laneHalf;
        int l = row0 + rr64;
        int s = c * 64 + tileCol + lane15;
        float ev = (liveScores && s <= l) ? __expf(0.125f * acc[i]) * aEr[rr64] : 0.f;
        float d  = (float)(l - s);
        float pv = aPr[rr64] * __expf(negA[rr64] * d * d);
        Fls[rr64 * LDSTR + tileCol + lane15] = (_Float16)(ev + pv);
      }
    }
    __syncthreads();

#pragma unroll
    for (int tt = 0; tt < 2; ++tt) {
      int dBase = q * 32 + tt * 16;
#pragma unroll
      for (int k0 = 0; k0 < 64; k0 += 32) {
        v16h a  = ld_a(Fls, 16 * r + lane15, k0, lane15, laneHalf);
        v16h bb = ld_bV(Vls, dBase, k0, lane15, laneHalf);
        outAcc[tt] = __builtin_amdgcn_wmma_f32_16x16x32_f16(false, a, false, bb,
                                                            (short)0, outAcc[tt],
                                                            false, false);
      }
    }
  }

  // ---- store O[b, l, h, d] (fp32) ----
#pragma unroll
  for (int tt = 0; tt < 2; ++tt) {
#pragma unroll
    for (int i = 0; i < 8; ++i) {
      int l    = row0 + 16 * r + i + 8 * laneHalf;
      int dcol = q * 32 + tt * 16 + lane15;
      Og[(((size_t)b * Lseq + l) * Hh + h) * 64 + dcol] = outAcc[tt][i];
    }
  }
}

extern "C" void kernel_launch(void* const* d_in, const int* in_sizes, int n_in,
                              void* d_out, int out_size, void* d_ws, size_t ws_size,
                              hipStream_t stream) {
  (void)in_sizes; (void)n_in; (void)out_size; (void)d_ws; (void)ws_size;
  const float* Q = (const float*)d_in[0];
  const float* K = (const float*)d_in[1];
  const float* V = (const float*)d_in[2];
  const float* S = (const float*)d_in[3];   // sigma [B,L,H]
  const float* G = (const float*)d_in[4];   // head_gate_logit [H]
  float* O = (float*)d_out;
  dim3 grid(Lseq / 64, Hh, 16);             // (row tiles, H, B)
  dim3 block(256);                          // 8 wave32s
  hipLaunchKernelGGL(anomaly_attn_kernel, grid, block, 0, stream, Q, K, V, S, G, O);
}
